// VectorQuantization_11879879543030
// MI455X (gfx1250) — matrix operand order repaired and checked by reading, not compiled, not run on hardware
//
#include <hip/hip_runtime.h>
#include <hip/hip_bf16.h>

// CDNA5 WMMA vector types
typedef __attribute__((ext_vector_type(16))) _Float16 v16h;
typedef __attribute__((ext_vector_type(8)))  float    v8f;

#define Hh   8
#define Dd   64
#define KC   512
#define HD   512          // h*d
#define TOKS_PER_BLOCK 128
#define THREADS 256       // 8 wave32 waves
#define NCT  (KC / 16)    // 32 cluster tiles

// ---------------------------------------------------------------------------
// Kernel 1: per-head distance GEMM via WMMA, argmin, and scatter-accumulate.
// grid = (bn/128, H), block = 256.
// ---------------------------------------------------------------------------
__global__ __launch_bounds__(THREADS)
void vq_assign_scatter(const float* __restrict__ x,
                       const float* __restrict__ means,
                       int*   __restrict__ out_ids,
                       float* __restrict__ ws_sum,   // [H, KC, Dd]
                       float* __restrict__ ws_cnt)   // [H, KC]
{
    __shared__ _Float16 smeans[KC * Dd];   // 64 KB: head codebook in f16
    __shared__ float    smnorm[KC];        // 2 KB: |m|^2 per cluster

    const int head = blockIdx.y;
    const int tid  = threadIdx.x;
    const float* mh = means + (size_t)head * KC * Dd;

    // Stage codebook into LDS (f16) and compute squared norms (fp32).
    for (int i = tid; i < KC * Dd; i += THREADS)
        smeans[i] = (_Float16)mh[i];
    for (int c = tid; c < KC; c += THREADS) {
        float s = 0.f;
        #pragma unroll
        for (int d = 0; d < Dd; ++d) { float v = mh[c * Dd + d]; s += v * v; }
        smnorm[c] = s;
    }
    __syncthreads();

    const int wave = tid >> 5;
    const int lane = tid & 31;
    const int hi   = lane >> 4;   // half-wave select
    const int lo   = lane & 15;   // row (A) / col (B,C) within 16
    const int tokbase = blockIdx.x * TOKS_PER_BLOCK + wave * 16;

    // A fragments (16x32 f16, this code's contiguous k order, same as B):
    //   M = lo; K = e + 16*hi
    v16h a0, a1;
    const float* xrow = x + (size_t)(tokbase + lo) * HD + head * Dd;
    #pragma unroll
    for (int e = 0; e < 16; ++e) {
        int k = e + 16 * hi;
        a0[e] = (_Float16)xrow[k];
        a1[e] = (_Float16)xrow[32 + k];
    }

    float bestv[8];
    int   besti[8];
    #pragma unroll
    for (int r = 0; r < 8; ++r) { bestv[r] = 3.4e38f; besti[r] = 0; }

    // Software-pipelined loop over 32 cluster tiles; K=64 = two 16x16x32
    // WMMAs. B fragments (32x16 f16): col = lo; K = e + 16*hi. Prefetch the
    // next tile's B regs + |m|^2 before the current WMMAs so the dscnt wait
    // overlaps matrix math instead of preceding it.
    v16h b0, b1;
    {
        const _Float16* mrow = &smeans[lo * Dd];
        #pragma unroll
        for (int e = 0; e < 16; ++e) {
            int k = e + 16 * hi;
            b0[e] = mrow[k];
            b1[e] = mrow[32 + k];
        }
    }
    float mn = smnorm[lo];

    for (int ct = 0; ct < NCT; ++ct) {
        v16h nb0, nb1;
        float nmn = 0.f;
        if (ct + 1 < NCT) {
            const _Float16* mrow = &smeans[((ct + 1) * 16 + lo) * Dd];
            #pragma unroll
            for (int e = 0; e < 16; ++e) {
                int k = e + 16 * hi;
                nb0[e] = mrow[k];
                nb1[e] = mrow[32 + k];
            }
            nmn = smnorm[(ct + 1) * 16 + lo];
        }

        v8f acc = {};
        acc = __builtin_amdgcn_wmma_f32_16x16x32_f16(false, a0, false, b0,
                                                     (short)0, acc, false, false);
        acc = __builtin_amdgcn_wmma_f32_16x16x32_f16(false, a1, false, b1,
                                                     (short)0, acc, false, false);

        const int cid = ct * 16 + lo;
        #pragma unroll
        for (int r = 0; r < 8; ++r) {
            // |x|^2 is constant per row -> irrelevant for argmin.
            float dist = mn - 2.0f * acc[r];
            if (dist < bestv[r]) { bestv[r] = dist; besti[r] = cid; }
        }

        b0 = nb0; b1 = nb1; mn = nmn;
    }

    // Cross-lane argmin over the 16 lanes of each half-wave (cols 0..15 of
    // each cluster tile live in separate lanes). xor masks 1..8 stay inside
    // each 16-lane half on wave32.
    #pragma unroll
    for (int m = 1; m <= 8; m <<= 1) {
        #pragma unroll
        for (int r = 0; r < 8; ++r) {
            float ov = __shfl_xor(bestv[r], m, 32);
            int   oi = __shfl_xor(besti[r], m, 32);
            if (ov < bestv[r] || (ov == bestv[r] && oi < besti[r])) {
                bestv[r] = ov; besti[r] = oi;
            }
        }
    }

    // C layout: lanes 0-15 hold rows 0-7 (VGPR r -> M=r), lanes 16-31 rows 8-15.
    if (lo == 0) {
        #pragma unroll
        for (int r = 0; r < 8; ++r) {
            int tok = tokbase + r + 8 * hi;
            out_ids[tok * Hh + head] = besti[r];
        }
    }

    // Scatter-accumulate: each half-wave owns 8 tokens; the 16 lanes split
    // the 64-dim vector (4 dims each). Counts by one lane per half.
    #pragma unroll
    for (int r = 0; r < 8; ++r) {
        int tok = tokbase + r + 8 * hi;
        int cid = besti[r];
        float*       dst = ws_sum + ((size_t)head * KC + cid) * Dd;
        const float* src = x + (size_t)tok * HD + head * Dd;
        #pragma unroll
        for (int j = 0; j < 4; ++j) {
            int d = lo + 16 * j;
            atomicAdd(&dst[d], src[d]);
        }
        if (lo == 0) atomicAdd(&ws_cnt[head * KC + cid], 1.0f);
    }
}

// ---------------------------------------------------------------------------
// Kernel 2: EMA finalize: (1-DECAY)*sum/(eps+cnt) + DECAY*means
// ---------------------------------------------------------------------------
__global__ void vq_finalize(const float* __restrict__ means,
                            const float* __restrict__ ws_sum,
                            const float* __restrict__ ws_cnt,
                            float* __restrict__ out_means, int total)
{
    int i = blockIdx.x * blockDim.x + threadIdx.x;
    if (i < total) {
        float cnt = ws_cnt[i >> 6];            // i / Dd
        float nm  = ws_sum[i] / (1e-6f + cnt);
        out_means[i] = 0.001f * nm + 0.999f * means[i];
    }
}

extern "C" void kernel_launch(void* const* d_in, const int* in_sizes, int n_in,
                              void* d_out, int out_size, void* d_ws, size_t ws_size,
                              hipStream_t stream)
{
    const float* x     = (const float*)d_in[0];   // [b, n, h*d] fp32
    const float* means = (const float*)d_in[1];   // [h, k, d] fp32

    const int bn = in_sizes[0] / HD;              // b*n = 16384

    // Outputs concatenated: cluster_ids [bn*H] (int32), then means [H*KC*Dd].
    int*   out_ids   = (int*)d_out;
    float* out_means = (float*)d_out + (size_t)bn * Hh;

    float* ws_sum = (float*)d_ws;                 // H*KC*Dd floats
    float* ws_cnt = ws_sum + (size_t)Hh * KC * Dd;

    hipMemsetAsync(d_ws, 0, ((size_t)Hh * KC * Dd + Hh * KC) * sizeof(float), stream);

    dim3 grid(bn / TOKS_PER_BLOCK, Hh);
    vq_assign_scatter<<<grid, THREADS, 0, stream>>>(x, means, out_ids, ws_sum, ws_cnt);

    int total = Hh * KC * Dd;
    vq_finalize<<<(total + 255) / 256, 256, 0, stream>>>(means, ws_sum, ws_cnt,
                                                         out_means, total);
}
